// VGTProModel_39281770889434
// MI455X (gfx1250) — compile-verified, matrix operand order checked by reading, not run on hardware
//
#include <hip/hip_runtime.h>

typedef __attribute__((ext_vector_type(16))) __bf16 v16bf;
typedef __attribute__((ext_vector_type(8)))  float  v8f;

#define HCH   128        // hidden channels
#define TCOLS 129        // digits + 1 (valid time columns)
#define NTILE 9          // ceil(144/16) time tiles
#define ROWS  152        // 4 halo + 144 + 4 halo
#define RSTR  132        // activation bf16 row stride (128 + 4 pad)
#define WSTR  130        // weight bf16 row stride (128 + 2 pad -> odd dword stride, conflict-free)
#define WELTS (3 * HCH * WSTR)          // 49,920 bf16
#define WBYTES (WELTS * 2)              // 99,840 bytes (multiple of 16)

// K-offset mapping for 16-bit A/B fragments (16x32 / 32x16 layout):
// VGPR p holds K = 2*(p&3) + (p>=4 ? 16 : 0) + hi*8  (pairs of consecutive K)
__device__ __forceinline__ int kmap(int p, int hi) {
    return 2 * (p & 3) + ((p & 4) ? 16 : 0) + hi * 8;
}

__global__ __launch_bounds__(256)
void prep_wbf(const float* __restrict__ conv_w, __bf16* __restrict__ wbf) {
    int i = blockIdx.x * 256 + threadIdx.x;          // over 3*128*128
    if (i < 3 * HCH * HCH) {
        int k   = i / (HCH * HCH);
        int rem = i - k * HCH * HCH;
        int m   = rem / HCH;
        int c   = rem - m * HCH;
        // conv_w is [O][C][K] (torch OIH); store padded [k][o][c] bf16 (LDS image)
        wbf[k * (HCH * WSTR) + m * WSTR + c] = (__bf16)conv_w[(m * HCH + c) * 3 + k];
    }
}

__global__ __launch_bounds__(256)
void conv_chain(const int* __restrict__ x, const float* __restrict__ emb_w,
                const float* __restrict__ red_w, const float* __restrict__ red_b,
                const __bf16* __restrict__ wbf, const float* __restrict__ conv_b,
                const float* __restrict__ out_w, const float* __restrict__ out_b,
                float* __restrict__ out)
{
    __shared__ __align__(16) unsigned int hbf_u[ROWS * RSTR / 2];  // bf16 acts, [t+4][c]
    __shared__ __align__(16) unsigned int wlds_u[WELTS / 2];       // bf16 weights [k][o][c]
    __shared__ float emb_s[10 * HCH];

    const int b    = blockIdx.x;
    const int tid  = threadIdx.x;
    const int lane = tid & 31;
    const int wave = tid >> 5;
    const int n    = lane & 15;   // column within tile (A: M row, B/C/D: N col)
    const int hi   = lane >> 4;   // K-half / M-half selector
    const int o0   = wave * 16;   // this wave's output-channel block

    __bf16* hbf = (__bf16*)hbf_u;

    // ---- async-copy the staged bf16 weights into LDS (gfx1250 async-to-LDS path) ----
    {
        unsigned lbase = (unsigned)(unsigned long long)(void*)wlds_u;  // LDS byte offset
        unsigned long long gbase = (unsigned long long)wbf;
        for (int off = tid * 16; off < WBYTES; off += 256 * 16) {
            unsigned la = lbase + (unsigned)off;
            unsigned long long ga = gbase + (unsigned long long)off;
            asm volatile("global_load_async_to_lds_b128 %0, %1, off"
                         :: "v"(la), "v"(ga) : "memory");
        }
    }

    // Zero activation LDS (halo rows + tail columns stay zero forever -> free padding)
    for (int i = tid; i < ROWS * RSTR / 2; i += 256) hbf_u[i] = 0u;
    for (int i = tid; i < 10 * HCH; i += 256) emb_s[i] = emb_w[i];

    asm volatile("s_wait_asynccnt 0" ::: "memory");
    __syncthreads();

    // ---------------- reducer init: h0 = relu(red_w @ cat + red_b), pad col 128 = 0 ----
    v8f hreg[NTILE];                       // fp32 residual stream, WMMA C/D layout
    int v1[NTILE], v2[NTILE];
    #pragma unroll
    for (int j = 0; j < NTILE; ++j) {
        int t = 16 * j + n;
        if (t < HCH) { v1[j] = x[b * 256 + t]; v2[j] = x[b * 256 + HCH + t]; }
        else         { v1[j] = 0;              v2[j] = 0; }
    }
    #pragma unroll
    for (int r = 0; r < 8; ++r) {
        int o = o0 + r + 8 * hi;
        float acc[NTILE];
        #pragma unroll
        for (int j = 0; j < NTILE; ++j) acc[j] = red_b[o];
        for (int c = 0; c < HCH; ++c) {
            float wlo = red_w[o * 2 * HCH + c];
            float whi = red_w[o * 2 * HCH + HCH + c];
            #pragma unroll
            for (int j = 0; j < NTILE; ++j)
                acc[j] += wlo * emb_s[v1[j] * HCH + c] + whi * emb_s[v2[j] * HCH + c];
        }
        #pragma unroll
        for (int j = 0; j < NTILE; ++j) {
            int t = 16 * j + n;
            float h0 = (t < HCH) ? fmaxf(acc[j], 0.f) : 0.f;   // col 128 is the zero pad
            hreg[j][r] = h0;
            if (t < TCOLS) hbf[(t + 4) * RSTR + o] = (__bf16)h0;
        }
    }
    float biasr[8];
    #pragma unroll
    for (int r = 0; r < 8; ++r) biasr[r] = conv_b[o0 + r + 8 * hi];
    __syncthreads();

    // ---------------- 131 residual dilated conv layers, pure LDS + WMMA ----------------
    const int m = o0 + n;                               // A-fragment row (output channel)

    for (int it = 0; it < 131; ++it) {
        const int d = (it < 4) ? 1 : (it < 8) ? 2 : 4;
        v8f acc[NTILE];
        #pragma unroll
        for (int j = 0; j < NTILE; ++j)
            acc[j] = (v8f){0.f, 0.f, 0.f, 0.f, 0.f, 0.f, 0.f, 0.f};

        for (int k = 0; k < 3; ++k) {                   // conv tap
            const int s = (k - 1) * d;                  // time shift
            for (int cb = 0; cb < 4; ++cb) {            // input-channel block of 32
                const int c0 = cb * 32;
                union { unsigned int u[8]; v16bf v; } A;
                #pragma unroll
                for (int p = 0; p < 8; ++p) {
                    int c = c0 + kmap(p, hi);
                    A.u[p] = wlds_u[(((k * HCH + m) * WSTR) + c) >> 1]; // weights from LDS
                }
                #pragma unroll
                for (int j = 0; j < NTILE; ++j) {
                    union { unsigned int u[8]; v16bf v; } Bm;
                    const int trow = 16 * j + n + s + 4; // halo guarantees in-bounds
                    #pragma unroll
                    for (int p = 0; p < 8; ++p) {
                        int c = c0 + kmap(p, hi);
                        Bm.u[p] = hbf_u[(trow * RSTR + c) >> 1];
                    }
                    acc[j] = __builtin_amdgcn_wmma_f32_16x16x32_bf16(
                        false, A.v, false, Bm.v, (short)0, acc[j], false, false);
                }
            }
        }
        __syncthreads();                                // all reads of hbf done
        #pragma unroll
        for (int j = 0; j < NTILE; ++j) {
            const int t = 16 * j + n;
            #pragma unroll
            for (int r = 0; r < 8; ++r)
                hreg[j][r] = fmaxf(acc[j][r] + biasr[r], 0.f) + hreg[j][r];
            if (t < TCOLS) {
                #pragma unroll
                for (int r = 0; r < 8; ++r)
                    hbf[(t + 4) * RSTR + (o0 + r + 8 * hi)] = (__bf16)hreg[j][r];
            }
        }
        __syncthreads();                                // writes visible to next layer
    }

    // ---------------- output projection: [10 x 128] @ h + out_b -----------------------
    for (int idx = tid; idx < TCOLS * 10; idx += 256) {
        int t = idx / 10, v = idx - 10 * t;
        float a = out_b[v];
        for (int c = 0; c < HCH; ++c)
            a += out_w[v * HCH + c] * (float)hbf[(t + 4) * RSTR + c];
        out[(b * TCOLS + t) * 10 + v] = a;
    }
}

extern "C" void kernel_launch(void* const* d_in, const int* in_sizes, int n_in,
                              void* d_out, int out_size, void* d_ws, size_t ws_size,
                              hipStream_t stream) {
    (void)in_sizes; (void)n_in; (void)out_size; (void)ws_size;
    const int*   x      = (const int*)  d_in[0];
    const float* emb_w  = (const float*)d_in[1];
    const float* red_w  = (const float*)d_in[2];
    const float* red_b  = (const float*)d_in[3];
    const float* conv_w = (const float*)d_in[4];
    const float* conv_b = (const float*)d_in[5];
    const float* out_w  = (const float*)d_in[6];
    const float* out_b  = (const float*)d_in[7];
    __bf16* wbf = (__bf16*)d_ws;                        // staged padded bf16 weights (~97.5 KB)

    prep_wbf<<<(3 * HCH * HCH + 255) / 256, 256, 0, stream>>>(conv_w, wbf);
    conv_chain<<<32, 256, 0, stream>>>(x, emb_w, red_w, red_b, wbf, conv_b,
                                       out_w, out_b, (float*)d_out);
}
